// graphDiffusionLayer_23622320128650
// MI455X (gfx1250) — matrix operand order repaired and checked by reading, not compiled, not run on hardware
//
#include <hip/hip_runtime.h>
#include <cstdint>

// ---------------- CDNA5 WMMA types ----------------
typedef __bf16 v16bf __attribute__((ext_vector_type(16)));
typedef float  v8f   __attribute__((ext_vector_type(8)));

#define C_IN   256
#define C_HID  512
#define ET     32                 // edges per tile (two 16-wide WMMA N subtiles)
#define GS2    (C_IN / 2 + 2)     // dword stride per edge column for g  (pair-packed K)
#define ZS2    (C_HID / 2 + 2)    // dword stride per edge column for zr

// ---- A fragment (16x32 bf16) from row-major global bf16, K pairs packed in dwords ----
// lanes 0-15: K 0..15 in u[0..7]; lanes 16-31: K 16..31
__device__ __forceinline__ v16bf frag_a_g(const uint32_t* W32, int ldw32, int row0, int kb, int lane) {
    const int m  = lane & 15;
    const int du = (lane >> 4) << 3;              // 0 or 8 dwords (K offset 0 or 16)
    const uint32_t* p = W32 + (size_t)(row0 + m) * ldw32 + (kb >> 1) + du;
    union { v16bf bf; uint32_t u[8]; } A;
#pragma unroll
    for (int v = 0; v < 4; ++v) A.u[v]     = p[v];
#pragma unroll
    for (int v = 0; v < 4; ++v) A.u[4 + v] = p[4 + v];
    return A.bf;
}

// ---- B fragment (32x16 bf16) from LDS tile stored column-major: [n][K/2] dwords ----
// lanes 0-15: K kb..kb+15; lanes 16-31: K kb+16..kb+31 -> 8 contiguous dwords per lane
__device__ __forceinline__ v16bf frag_b_lds(const uint32_t* gcol, int stride2, int kb, int ncol0, int lane) {
    const int n    = ncol0 + (lane & 15);
    const int row0 = (kb >> 1) + ((lane >> 4) << 3);
    const uint32_t* p = gcol + n * stride2 + row0;
    union { v16bf bf; uint32_t u[8]; } B;
#pragma unroll
    for (int v = 0; v < 8; ++v) B.u[v] = p[v];    // -> paired ds_load_b64/b128
    return B.bf;
}

// ---------------- Kernel A: prep (zero out/stats, W -> bf16 + transpose) ----------------
__global__ void k_prep(const float* __restrict__ W, __bf16* __restrict__ Wb,
                       __bf16* __restrict__ Wt, float* __restrict__ stats,
                       float* __restrict__ out, int outN) {
    const int i = blockIdx.x * blockDim.x + threadIdx.x;
    if (i < outN) out[i] = 0.0f;
    if (i < C_HID * C_IN) {
        const float v = W[i];
        Wb[i] = (__bf16)v;
        const int h = i >> 8;
        const int c = i & (C_IN - 1);
        Wt[c * C_HID + h] = (__bf16)v;
    }
    if (i < 2 * C_HID) stats[i] = 0.0f;   // sum[0..511], sumsq[512..1023]
}

// ---------------- Kernel B: gather + GEMM1 (WMMA) + mean/var stats ----------------
__global__ void __launch_bounds__(256)
k_stats(const float* __restrict__ x, const __bf16* __restrict__ Wb,
        const int* __restrict__ iInd, const int* __restrict__ jInd,
        float* __restrict__ gsum, float* __restrict__ gsq,
        int E, int Nn, int nTiles) {
    __shared__ int      ii_s[ET], jj_s[ET];
    __shared__ uint32_t g_s[ET * GS2];              // 256x32 bf16, column-major pair-packed
    __shared__ float2   red[8][16][18];             // per-wave transpose scratch (value, value^2)
    __shared__ float    lsum[C_HID], lsq[C_HID];

    const int tid  = threadIdx.x;
    const int w    = tid >> 5;
    const int lane = tid & 31;

    for (int h = tid; h < C_HID; h += 256) { lsum[h] = 0.0f; lsq[h] = 0.0f; }

    const int et   = w & 1;
    const int hgrp = w >> 1;

    size_t woff = 0;   // laundered offset: defeats LICM while keeping global addrspace

    for (int tile = blockIdx.x; tile < nTiles; tile += gridDim.x) {
        asm volatile("" : "+s"(woff));
        const uint32_t* Wb32 = (const uint32_t*)Wb + woff;   // ld = 128 dwords

        const int e0 = tile * ET;
        __syncthreads();                            // previous iter done / init visible
        if (tid < ET) {
            const int e = e0 + tid;
            ii_s[tid] = (e < E) ? iInd[e] : 0;
            jj_s[tid] = (e < E) ? jInd[e] : 0;
        }
        __syncthreads();
        {   // gather: thread -> (edge el, 32 channels); column-major pair-packed store
            const int  el = tid & (ET - 1);
            const int  cg = tid >> 5;
            const int  ii = ii_s[el], jj = jj_s[el];
            const bool valid = (e0 + el) < E;
            __bf16* gb = (__bf16*)g_s;
#pragma unroll 4
            for (int cc = 0; cc < 32; ++cc) {
                const int c = (cg << 5) + cc;
                const float v = valid ? (x[c * Nn + ii] - x[c * Nn + jj]) : 0.0f;
                gb[((el * GS2 + (c >> 1)) << 1) | (c & 1)] = (__bf16)v;
            }
        }
        __syncthreads();

        // GEMM1: k outer (kept rolled to bound VGPR pressure), 8 independent accumulators
        v8f acc[8] = {};
#pragma unroll 1
        for (int kb = 0; kb < C_IN; kb += 32) {
            v16bf b = frag_b_lds(g_s, GS2, kb, et << 4, lane);
#pragma unroll
            for (int i = 0; i < 8; ++i) {
                v16bf a = frag_a_g(Wb32, C_IN / 2, (((hgrp << 3) + i) << 4), kb, lane);
                acc[i] = __builtin_amdgcn_wmma_f32_16x16x32_bf16(false, a, false, b,
                                                                 (short)0, acc[i], false, false);
            }
        }

        // per-tile row sums via per-wave LDS transpose (same-wave DS ops are in-order)
        for (int i = 0; i < 8; ++i) {
            const int h0 = (((hgrp << 3) + i) << 4);
#pragma unroll
            for (int r = 0; r < 8; ++r) {
                const int m = r + ((lane >> 4) << 3);
                red[w][m][lane & 15] = make_float2(acc[i][r], acc[i][r] * acc[i][r]);
            }
            const int row = lane & 15;
            const int cb  = (lane >> 4) << 3;
            float s = 0.0f, q = 0.0f;
#pragma unroll
            for (int nn = 0; nn < 8; ++nn) {
                float2 t = red[w][row][cb + nn];
                s += t.x; q += t.y;
            }
            s += __shfl_xor(s, 16, 32);
            q += __shfl_xor(q, 16, 32);
            if (lane < 16) {
                atomicAdd(&lsum[h0 + row], s);      // 16 lanes, distinct banks
                atomicAdd(&lsq[h0 + row], q);
            }
        }
    }
    __syncthreads();
    for (int h = tid; h < C_HID; h += 256) {
        atomicAdd(&gsum[h], lsum[h]);
        atomicAdd(&gsq[h], lsq[h]);
    }
}

// ---------------- Kernel C: finalize instance-norm scale/shift ----------------
__global__ void k_finalize(float* __restrict__ stats, float invE) {
    const int h = blockIdx.x * blockDim.x + threadIdx.x;
    if (h < C_HID) {
        const float mu  = stats[h] * invE;
        const float var = stats[C_HID + h] * invE - mu * mu;
        const float sc  = rsqrtf(fmaxf(var, 0.0f) + 1e-5f);
        stats[2 * C_HID + h] = sc;          // scale
        stats[3 * C_HID + h] = -mu * sc;    // shift
    }
}

// ---------------- Kernel D: gather + GEMM1 + norm/ReLU + GEMM2 + scatter ----------------
__global__ void __launch_bounds__(256)
k_main(const float* __restrict__ x, const __bf16* __restrict__ Wb,
       const __bf16* __restrict__ Wt, const int* __restrict__ iInd,
       const int* __restrict__ jInd, const float* __restrict__ scale_g,
       const float* __restrict__ shift_g, float* __restrict__ out,
       int E, int Nn) {
    __shared__ int      ii_s[ET], jj_s[ET];
    __shared__ uint32_t g_s[ET * GS2];     // 256x32 bf16 gathered tile, column-major
    __shared__ uint32_t zr_s[ET * ZS2];    // 512x32 bf16 normalized tile, column-major
    __shared__ float    sc_s[C_HID], sh_s[C_HID];

    const int tid  = threadIdx.x;
    const int w    = tid >> 5;
    const int lane = tid & 31;
    const uint32_t* Wb32 = (const uint32_t*)Wb;      // ld 128 dwords
    const uint32_t* Wt32 = (const uint32_t*)Wt;      // ld 256 dwords
    const int e0 = blockIdx.x * ET;

    // warm Wt (L2 resident; gfx1250 prefetch path)
    __builtin_prefetch((const char*)Wt + tid * 1024, 0, 0);

    for (int h = tid; h < C_HID; h += 256) { sc_s[h] = scale_g[h]; sh_s[h] = shift_g[h]; }
    if (tid < ET) {
        const int e = e0 + tid;
        ii_s[tid] = (e < E) ? iInd[e] : 0;
        jj_s[tid] = (e < E) ? jInd[e] : 0;
    }
    __syncthreads();
    {   // gather
        const int  el = tid & (ET - 1);
        const int  cg = tid >> 5;
        const int  ii = ii_s[el], jj = jj_s[el];
        const bool valid = (e0 + el) < E;
        __bf16* gb = (__bf16*)g_s;
#pragma unroll 4
        for (int cc = 0; cc < 32; ++cc) {
            const int c = (cg << 5) + cc;
            const float v = valid ? (x[c * Nn + ii] - x[c * Nn + jj]) : 0.0f;
            gb[((el * GS2 + (c >> 1)) << 1) | (c & 1)] = (__bf16)v;
        }
    }
    __syncthreads();

    const int et = w & 1;
    // GEMM1 (k outer rolled, acc[8]) + fused instance-norm affine + ReLU -> zr tile
    {
        const int hgrp = w >> 1;
        v8f acc[8] = {};
#pragma unroll 1
        for (int kb = 0; kb < C_IN; kb += 32) {
            v16bf b = frag_b_lds(g_s, GS2, kb, et << 4, lane);
#pragma unroll
            for (int i = 0; i < 8; ++i) {
                v16bf a = frag_a_g(Wb32, C_IN / 2, (((hgrp << 3) + i) << 4), kb, lane);
                acc[i] = __builtin_amdgcn_wmma_f32_16x16x32_bf16(false, a, false, b,
                                                                 (short)0, acc[i], false, false);
            }
        }
        __bf16* zb = (__bf16*)zr_s;
        const int col = (et << 4) + (lane & 15);
#pragma unroll
        for (int i = 0; i < 8; ++i) {
            const int h0 = (((hgrp << 3) + i) << 4);
#pragma unroll
            for (int r = 0; r < 8; ++r) {
                const int h = h0 + r + ((lane >> 4) << 3);
                const float v = fmaxf(fmaf(acc[i][r], sc_s[h], sh_s[h]), 0.0f);
                zb[((col * ZS2 + (h >> 1)) << 1) | (h & 1)] = (__bf16)v;
            }
        }
    }
    __syncthreads();
    // GEMM2 (k outer rolled, acc[4]): out2 = Wt (256x512) x zr (512x32), then scatter +-
    {
        const int cgrp = w >> 1;
        v8f acc[4] = {};
#pragma unroll 1
        for (int kb = 0; kb < C_HID; kb += 32) {
            v16bf b = frag_b_lds(zr_s, ZS2, kb, et << 4, lane);
#pragma unroll
            for (int i = 0; i < 4; ++i) {
                v16bf a = frag_a_g(Wt32, C_HID / 2, (((cgrp << 2) + i) << 4), kb, lane);
                acc[i] = __builtin_amdgcn_wmma_f32_16x16x32_bf16(false, a, false, b,
                                                                 (short)0, acc[i], false, false);
            }
        }
        const int n  = lane & 15;
        const int el = (et << 4) + n;
        if (e0 + el < E) {
            const int ii = ii_s[el], jj = jj_s[el];
#pragma unroll
            for (int i = 0; i < 4; ++i) {
                const int c0 = (((cgrp << 2) + i) << 4);
#pragma unroll
                for (int r = 0; r < 8; ++r) {
                    const int c = c0 + r + ((lane >> 4) << 3);
                    const float v = acc[i][r];
                    atomicAdd(&out[(size_t)c * Nn + ii],  v);
                    atomicAdd(&out[(size_t)c * Nn + jj], -v);
                }
            }
        }
    }
}

extern "C" void kernel_launch(void* const* d_in, const int* in_sizes, int n_in,
                              void* d_out, int out_size, void* d_ws, size_t ws_size,
                              hipStream_t stream) {
    const float* x    = (const float*)d_in[0];
    const float* W    = (const float*)d_in[1];
    const int*   iInd = (const int*)d_in[2];
    const int*   jInd = (const int*)d_in[3];
    float*       out  = (float*)d_out;

    const int xN = in_sizes[0];            // C_IN * N_NODES
    const int E  = in_sizes[2];            // edges
    const int Nn = xN / C_IN;              // nodes

    // workspace layout: Wb bf16 [512x256], Wt bf16 [256x512], stats f32 [4*512]
    __bf16* Wb    = (__bf16*)d_ws;
    __bf16* Wt    = (__bf16*)((char*)d_ws + (size_t)C_HID * C_IN * 2);
    float*  stats = (float*)((char*)d_ws + (size_t)2 * C_HID * C_IN * 2);

    const int nTiles = (E + ET - 1) / ET;

    int prepN = xN > C_HID * C_IN ? xN : C_HID * C_IN;
    k_prep<<<(prepN + 255) / 256, 256, 0, stream>>>(W, Wb, Wt, stats, out, xN);

    int statsBlocks = nTiles < 640 ? nTiles : 640;
    k_stats<<<statsBlocks, 256, 0, stream>>>(x, Wb, iInd, jInd,
                                             stats, stats + C_HID, E, Nn, nTiles);

    k_finalize<<<(C_HID + 255) / 256, 256, 0, stream>>>(stats, 1.0f / (float)E);

    k_main<<<nTiles, 256, 0, stream>>>(x, Wb, Wt, iInd, jInd,
                                       stats + 2 * C_HID, stats + 3 * C_HID,
                                       out, E, Nn);
}
